// NeuralMemory_12919261626918
// MI455X (gfx1250) — matrix operand order repaired and checked by reading, not compiled, not run on hardware
//
#include <hip/hip_runtime.h>

#define D_IN   512
#define HID    1024          // 2*D_IN
#define SEQLEN 4096
#define CHUNK  256
#define NCHUNK 16
#define BATCH  4
#define ROWS   (BATCH*CHUNK)        // 1024 rows per chunk step
#define TOTALE (BATCH*CHUNK*D_IN)   // elements in the loss mean = 524288

typedef __attribute__((ext_vector_type(16))) __bf16 v16bf;
typedef __attribute__((ext_vector_type(8)))  __bf16 v8bf;
typedef __attribute__((ext_vector_type(8)))  float  v8f;

// ---------------------------------------------------------------- helpers
__device__ __forceinline__ unsigned short f32_to_bf16_rne(float f) {
  union { float f; unsigned u; } v; v.f = f;
  unsigned u = v.u;
  u += 0x7FFFu + ((u >> 16) & 1u);          // round-to-nearest-even
  return (unsigned short)(u >> 16);
}

__device__ __forceinline__ float gelu_exact(float x) {
  return 0.5f * x * (1.0f + erff(x * 0.70710678118654752f));
}
__device__ __forceinline__ float gelu_grad(float x) {
  return 0.5f * (1.0f + erff(x * 0.70710678118654752f))
       + x * 0.39894228040143268f * __expf(-0.5f * x * x);
}

// A fragment (16x32 bf16, ISA §7.12.2): lanes 0-15 -> M=lane, K=k0+{0..7,16..23}
//                                       lanes 16-31 -> M=lane-16, K=k0+{8..15,24..31}
__device__ __forceinline__ v16bf load_frag_a(const __bf16* row, int k0, int half) {
  const int kLo = k0 + half * 8;
  const int kHi = k0 + 16 + half * 8;
  v8bf lo = *(const v8bf*)(row + kLo);
  v8bf hi = *(const v8bf*)(row + kHi);
  v16bf r;
#pragma unroll
  for (int i = 0; i < 8; ++i) { r[i] = lo[i]; r[i + 8] = hi[i]; }
  return r;
}

// B fragment (32x16 bf16, column-major per column N=lane&15):
// lanes 0-15 hold K=k0+{0..7,16..23}, lanes 16-31 hold K=k0+{8..15,24..31} (matches A)
__device__ __forceinline__ v16bf load_frag_b(const __bf16* col, int k0, int half) {
  const int kBase = k0 + half * 8;
  v8bf lo = *(const v8bf*)(col + kBase);
  v8bf hi = *(const v8bf*)(col + kBase + 16);
  v16bf r;
#pragma unroll
  for (int i = 0; i < 8; ++i) { r[i] = lo[i]; r[i + 8] = hi[i]; }
  return r;
}

// ---------------------------------------------------------------- GEMM (WMMA)
// C[M,N] (f32, row-major) = A[M,K] (bf16, rm) x Bt[N,K]^T (bf16, rm)
// Optional fused bf16 copy of C into Cbf (pass nullptr to skip).
// grid = (N/64, M/32): one wave computes a 32x64 tile (2x4 WMMA fragments,
// 8 v_wmma per K-step vs 12 b128 loads; B fragments feed two WMMAs each).
__global__ __launch_bounds__(32) void gemm_bf16_wmma(
    float* __restrict__ C, unsigned short* __restrict__ Cbf,
    const __bf16* __restrict__ A, const __bf16* __restrict__ Bt,
    int M, int N, int K)
{
  const int lane  = threadIdx.x & 31;
  const int half  = lane >> 4;
  const int r     = lane & 15;
  const int tileM = blockIdx.y << 5;
  const int tileN = blockIdx.x << 6;

  v8f acc[2][4] = {};
  const __bf16* Arow0 = A  + (size_t)(tileM + r) * K;
  const __bf16* Arow1 = Arow0 + (size_t)16 * K;
  const __bf16* B0    = Bt + (size_t)(tileN + r) * K;
  const __bf16* B1    = B0 + (size_t)16 * K;
  const __bf16* B2    = B0 + (size_t)32 * K;
  const __bf16* B3    = B0 + (size_t)48 * K;

  for (int k0 = 0; k0 < K; k0 += 32) {
    v16bf a0 = load_frag_a(Arow0, k0, half);
    v16bf a1 = load_frag_a(Arow1, k0, half);
    v16bf b0 = load_frag_b(B0, k0, half);
    v16bf b1 = load_frag_b(B1, k0, half);
    v16bf b2 = load_frag_b(B2, k0, half);
    v16bf b3 = load_frag_b(B3, k0, half);
    acc[0][0] = __builtin_amdgcn_wmma_f32_16x16x32_bf16(false, a0, false, b0, (short)0, acc[0][0], false, false);
    acc[1][0] = __builtin_amdgcn_wmma_f32_16x16x32_bf16(false, a1, false, b0, (short)0, acc[1][0], false, false);
    acc[0][1] = __builtin_amdgcn_wmma_f32_16x16x32_bf16(false, a0, false, b1, (short)0, acc[0][1], false, false);
    acc[1][1] = __builtin_amdgcn_wmma_f32_16x16x32_bf16(false, a1, false, b1, (short)0, acc[1][1], false, false);
    acc[0][2] = __builtin_amdgcn_wmma_f32_16x16x32_bf16(false, a0, false, b2, (short)0, acc[0][2], false, false);
    acc[1][2] = __builtin_amdgcn_wmma_f32_16x16x32_bf16(false, a1, false, b2, (short)0, acc[1][2], false, false);
    acc[0][3] = __builtin_amdgcn_wmma_f32_16x16x32_bf16(false, a0, false, b3, (short)0, acc[0][3], false, false);
    acc[1][3] = __builtin_amdgcn_wmma_f32_16x16x32_bf16(false, a1, false, b3, (short)0, acc[1][3], false, false);
  }

  // C/D layout: VGPR g -> row sub-tile row g (lanes 0-15) / g+8 (lanes 16-31),
  // col = tileN + q*16 + (lane&15)
#pragma unroll
  for (int p = 0; p < 2; ++p) {
#pragma unroll
    for (int g = 0; g < 8; ++g) {
      const size_t row = (size_t)(tileM + p * 16 + g + half * 8);
#pragma unroll
      for (int q = 0; q < 4; ++q) {
        const size_t idx = row * N + tileN + q * 16 + r;
        const float val = acc[p][q][g];
        C[idx] = val;
        if (Cbf) Cbf[idx] = f32_to_bf16_rne(val);
      }
    }
  }
}

// ---------------------------------------------------------------- elementwise
__global__ __launch_bounds__(256) void k_init_state(
    float* __restrict__ W, float* __restrict__ S, const float* __restrict__ W0, int n)
{
  int i = blockIdx.x * 256 + threadIdx.x;
  if (i < n) { W[i] = W0[i]; S[i] = 0.0f; }
}

__global__ __launch_bounds__(256) void k_convert_bf16(
    unsigned short* __restrict__ o, const float* __restrict__ in, int n)
{
  int i = blockIdx.x * 256 + threadIdx.x;
  if (i < n) o[i] = f32_to_bf16_rne(in[i]);
}

// out[N,M] (bf16) = transpose(in[M,N] f32)
__global__ __launch_bounds__(256) void k_transpose_bf16(
    unsigned short* __restrict__ o, const float* __restrict__ in, int M, int N)
{
  int idx = blockIdx.x * 256 + threadIdx.x;
  if (idx >= M * N) return;
  int i = idx / N, j = idx % N;
  o[(size_t)j * M + i] = f32_to_bf16_rne(in[(size_t)i * N + j]);
}

// gather chunk c of x ([B,SEQ,D] f32) into contiguous [ROWS,D] bf16
__global__ __launch_bounds__(256) void k_gather_chunk_bf16(
    unsigned short* __restrict__ o, const float* __restrict__ x, int c)
{
  int idx = blockIdx.x * 256 + threadIdx.x;
  if (idx >= ROWS * D_IN) return;
  int r = idx / D_IN, col = idx % D_IN;
  int b = r / CHUNK, t = r % CHUNK;
  size_t g = ((size_t)b * SEQLEN + (size_t)c * CHUNK + t) * D_IN + col;
  o[idx] = f32_to_bf16_rne(x[g]);
}

// scatter chunk c of contiguous [ROWS,D] f32 into out [B,SEQ,D]
__global__ __launch_bounds__(256) void k_scatter_chunk(
    float* __restrict__ out, const float* __restrict__ p, int c)
{
  int idx = blockIdx.x * 256 + threadIdx.x;
  if (idx >= ROWS * D_IN) return;
  int r = idx / D_IN, col = idx % D_IN;
  int b = r / CHUNK, t = r % CHUNK;
  size_t g = ((size_t)b * SEQLEN + (size_t)c * CHUNK + t) * D_IN + col;
  out[g] = p[idx];
}

// a = gelu(h): fused f32 + bf16 outputs
__global__ __launch_bounds__(256) void k_gelu_convert(
    float* __restrict__ a, unsigned short* __restrict__ abf,
    const float* __restrict__ h, int n)
{
  int i = blockIdx.x * 256 + threadIdx.x;
  if (i < n) { float v = gelu_exact(h[i]); a[i] = v; abf[i] = f32_to_bf16_rne(v); }
}

// dP = (2/TOTALE) * (P - V): fused f32 + bf16 outputs
__global__ __launch_bounds__(256) void k_dpred_convert(
    float* __restrict__ dp, unsigned short* __restrict__ dpbf,
    const float* __restrict__ p, const float* __restrict__ v, int n)
{
  int i = blockIdx.x * 256 + threadIdx.x;
  if (i < n) {
    float d = (p[i] - v[i]) * (2.0f / (float)TOTALE);
    dp[i] = d; dpbf[i] = f32_to_bf16_rne(d);
  }
}

// dH = dA * gelu'(H), in place on dA
__global__ __launch_bounds__(256) void k_dgelu_inplace(
    float* __restrict__ da, const float* __restrict__ h, int n)
{
  int i = blockIdx.x * 256 + threadIdx.x;
  if (i < n) da[i] *= gelu_grad(h[i]);
}

// if (*upd): S = decay*S - lr*G ; W = (1-alpha)*W + S
__global__ __launch_bounds__(256) void k_update(
    float* __restrict__ W, float* __restrict__ S, const float* __restrict__ G, int n,
    const float* __restrict__ alpha_t, const float* __restrict__ lr_t,
    const float* __restrict__ decay_t, const int* __restrict__ upd)
{
  if (*upd == 0) return;
  int i = blockIdx.x * 256 + threadIdx.x;
  if (i >= n) return;
  float alpha = 1.0f / (1.0f + __expf(-*alpha_t));
  float lr    = 1.0f / (1.0f + __expf(-*lr_t));
  float decay = 1.0f / (1.0f + __expf(-*decay_t));
  float s = decay * S[i] - lr * G[i];
  S[i] = s;
  W[i] = (1.0f - alpha) * W[i] + s;
}

// ---------------------------------------------------------------- host side
static inline void gemm(hipStream_t s, float* C, unsigned short* Cbf,
                        const void* A, const void* Bt, int M, int N, int K) {
  dim3 g((unsigned)(N / 64), (unsigned)(M / 32));
  gemm_bf16_wmma<<<g, 32, 0, s>>>(C, Cbf, (const __bf16*)A, (const __bf16*)Bt, M, N, K);
}
static inline dim3 ew(int n) { return dim3((unsigned)((n + 255) / 256)); }

extern "C" void kernel_launch(void* const* d_in, const int* in_sizes, int n_in,
                              void* d_out, int out_size, void* d_ws, size_t ws_size,
                              hipStream_t stream) {
  const float* x       = (const float*)d_in[0];
  const float* w_q     = (const float*)d_in[1];
  const float* w_k     = (const float*)d_in[2];
  const float* w_v     = (const float*)d_in[3];
  const float* mem_w1  = (const float*)d_in[4];
  const float* mem_w2  = (const float*)d_in[5];
  const float* alpha_t = (const float*)d_in[6];
  const float* lr_t    = (const float*)d_in[7];
  const float* decay_t = (const float*)d_in[8];
  const int*   upd     = (const int*)d_in[9];
  float* out = (float*)d_out;

  char* ws = (char*)d_ws;
  size_t off = 0;
  auto alloc = [&](size_t bytes) -> void* {
    void* p = ws + off;
    off = (off + bytes + 255) & ~(size_t)255;
    return p;
  };

  const int NW = D_IN * HID;              // 524288 (|W1| == |W2|)
  const int NH = ROWS * HID;              // 1024*1024
  const int ND = ROWS * D_IN;             // 1024*512

  // persistent f32 state
  float* W1 = (float*)alloc((size_t)NW * 4);
  float* W2 = (float*)alloc((size_t)NW * 4);
  float* S1 = (float*)alloc((size_t)NW * 4);
  float* S2 = (float*)alloc((size_t)NW * 4);
  // static bf16 projections (transposed for GEMM Bt operand)
  unsigned short* wq_t = (unsigned short*)alloc((size_t)D_IN * D_IN * 2);
  unsigned short* wk_t = (unsigned short*)alloc((size_t)D_IN * D_IN * 2);
  unsigned short* wv_t = (unsigned short*)alloc((size_t)D_IN * D_IN * 2);
  // per-chunk buffers
  unsigned short* xc_bf  = (unsigned short*)alloc((size_t)ND * 2);
  float*          Kc     = (float*)alloc((size_t)ND * 4);   // reused as Q in read-out
  float*          Vc     = (float*)alloc((size_t)ND * 4);
  unsigned short* Kc_bf  = (unsigned short*)alloc((size_t)ND * 2);
  unsigned short* Kct_bf = (unsigned short*)alloc((size_t)ND * 2);
  unsigned short* W1t_bf = (unsigned short*)alloc((size_t)NW * 2);
  unsigned short* W2t_bf = (unsigned short*)alloc((size_t)NW * 2);
  unsigned short* W2_bf  = (unsigned short*)alloc((size_t)NW * 2);
  float*          H      = (float*)alloc((size_t)NH * 4);
  float*          Af     = (float*)alloc((size_t)NH * 4);
  unsigned short* A_bf   = (unsigned short*)alloc((size_t)NH * 2);
  unsigned short* At_bf  = (unsigned short*)alloc((size_t)NH * 2);
  float*          P      = (float*)alloc((size_t)ND * 4);
  float*          dP     = (float*)alloc((size_t)ND * 4);
  unsigned short* dP_bf  = (unsigned short*)alloc((size_t)ND * 2);
  unsigned short* dPt_bf = (unsigned short*)alloc((size_t)ND * 2);
  float*          dA     = (float*)alloc((size_t)NH * 4);
  unsigned short* dHt_bf = (unsigned short*)alloc((size_t)NH * 2);
  float*          G1     = (float*)alloc((size_t)NW * 4);
  float*          G2     = (float*)alloc((size_t)NW * 4);

  // ---- setup: copy master weights, zero momentum, cast projections
  k_init_state<<<ew(NW), 256, 0, stream>>>(W1, S1, mem_w1, NW);
  k_init_state<<<ew(NW), 256, 0, stream>>>(W2, S2, mem_w2, NW);
  k_transpose_bf16<<<ew(D_IN * D_IN), 256, 0, stream>>>(wq_t, w_q, D_IN, D_IN);
  k_transpose_bf16<<<ew(D_IN * D_IN), 256, 0, stream>>>(wk_t, w_k, D_IN, D_IN);
  k_transpose_bf16<<<ew(D_IN * D_IN), 256, 0, stream>>>(wv_t, w_v, D_IN, D_IN);

  // ---- sequential memory-update scan over 16 chunks
  for (int c = 0; c < NCHUNK; ++c) {
    k_gather_chunk_bf16<<<ew(ND), 256, 0, stream>>>(xc_bf, x, c);
    gemm(stream, Kc, Kc_bf, xc_bf, wk_t, ROWS, D_IN, D_IN);      // k = xc @ w_k (+bf16)
    gemm(stream, Vc, nullptr, xc_bf, wv_t, ROWS, D_IN, D_IN);    // v = xc @ w_v
    k_transpose_bf16<<<ew(ND), 256, 0, stream>>>(Kct_bf, Kc, ROWS, D_IN);
    k_transpose_bf16<<<ew(NW), 256, 0, stream>>>(W1t_bf, W1, D_IN, HID);
    gemm(stream, H, nullptr, Kc_bf, W1t_bf, ROWS, HID, D_IN);    // h = k @ W1
    k_gelu_convert<<<ew(NH), 256, 0, stream>>>(Af, A_bf, H, NH); // a = gelu(h) (+bf16)
    k_transpose_bf16<<<ew(NH), 256, 0, stream>>>(At_bf, Af, ROWS, HID);
    k_transpose_bf16<<<ew(NW), 256, 0, stream>>>(W2t_bf, W2, HID, D_IN);
    k_convert_bf16<<<ew(NW), 256, 0, stream>>>(W2_bf, W2, NW);
    gemm(stream, P, nullptr, A_bf, W2t_bf, ROWS, D_IN, HID);     // pred = a @ W2
    k_dpred_convert<<<ew(ND), 256, 0, stream>>>(dP, dP_bf, P, Vc, ND);
    k_transpose_bf16<<<ew(ND), 256, 0, stream>>>(dPt_bf, dP, ROWS, D_IN);
    gemm(stream, G2, nullptr, At_bf, dPt_bf, HID, D_IN, ROWS);   // g2 = a^T @ dpred
    gemm(stream, dA, nullptr, dP_bf, W2_bf, ROWS, HID, D_IN);    // da = dpred @ W2^T
    k_dgelu_inplace<<<ew(NH), 256, 0, stream>>>(dA, H, NH);      // dh = da * gelu'(h)
    k_transpose_bf16<<<ew(NH), 256, 0, stream>>>(dHt_bf, dA, ROWS, HID);
    gemm(stream, G1, nullptr, Kct_bf, dHt_bf, D_IN, HID, ROWS);  // g1 = k^T @ dh
    k_update<<<ew(NW), 256, 0, stream>>>(W1, S1, G1, NW, alpha_t, lr_t, decay_t, upd);
    k_update<<<ew(NW), 256, 0, stream>>>(W2, S2, G2, NW, alpha_t, lr_t, decay_t, upd);
  }

  // ---- read-out: out = gelu(x @ w_q @ W1) @ W2, processed in 16 row blocks
  k_transpose_bf16<<<ew(NW), 256, 0, stream>>>(W1t_bf, W1, D_IN, HID);
  k_transpose_bf16<<<ew(NW), 256, 0, stream>>>(W2t_bf, W2, HID, D_IN);
  for (int c = 0; c < NCHUNK; ++c) {
    k_gather_chunk_bf16<<<ew(ND), 256, 0, stream>>>(xc_bf, x, c);
    gemm(stream, Kc, Kc_bf, xc_bf, wq_t, ROWS, D_IN, D_IN);      // q (+bf16)
    gemm(stream, H, nullptr, Kc_bf, W1t_bf, ROWS, HID, D_IN);
    k_gelu_convert<<<ew(NH), 256, 0, stream>>>(Af, A_bf, H, NH);
    gemm(stream, P, nullptr, A_bf, W2t_bf, ROWS, D_IN, HID);
    k_scatter_chunk<<<ew(ND), 256, 0, stream>>>(out, P, c);
  }
  (void)in_sizes; (void)n_in; (void)out_size; (void)ws_size;
}
